// _IPEXGatedMLPMOECPU_59227599011939
// MI455X (gfx1250) — compile-verified, matrix operand order checked by reading, not run on hardware
//
#include <hip/hip_runtime.h>

// Problem constants (from reference)
#define E_ 8
#define I_ 2816
#define H_ 1024
#define T_ 8192
#define TOPK_ 2

// Tiling
#define TM_ 32           // pair-rows per block tile
#define TN_ 128          // output cols per block tile (8 wmma tiles wide)
#define TK_ 32           // K step (one bf16 wmma K)
#define NPAIR_ (T_ * TOPK_)            // 16384 routed pairs exactly
#define CAP_   (NPAIR_ + E_ * TM_)     // tile-aligned capacity
#define MAX_TILES_ (NPAIR_ / TM_ + E_) // 520 worst-case row tiles

typedef __attribute__((ext_vector_type(8)))  float  v8f;
typedef __attribute__((ext_vector_type(16))) __bf16 v16bf;
typedef __attribute__((ext_vector_type(8)))  __bf16 bf16x8;
typedef __attribute__((ext_vector_type(4)))  __bf16 bf16x4;

union FragBF { v16bf v; bf16x8 h[2]; };

// Load one 16x32 bf16 operand fragment (A or B) from a row-major [rows][TK_] LDS tile.
// Wave32 layout (ISA 7.12.2): lane&15 = row, lane>>4 selects K-half:
//   kb=0 -> K 0..7 (V0-3) and 16..23 (V4-7); kb=8 -> K 8..15 and 24..31.
__device__ __forceinline__ v16bf load_frag(const __bf16* tile, int row, int kb) {
    FragBF f;
    f.h[0] = *(const bf16x8*)(tile + row * TK_ + kb);
    f.h[1] = *(const bf16x8*)(tile + row * TK_ + kb + 16);
    return f.v;
}

__device__ __forceinline__ bf16x4 cvt4(float4 v) {
    bf16x4 p;
    p[0] = (__bf16)v.x; p[1] = (__bf16)v.y;
    p[2] = (__bf16)v.z; p[3] = (__bf16)v.w;
    return p;
}

#define WMMA_BF16(A, B, C) \
    __builtin_amdgcn_wmma_f32_16x16x32_bf16(false, (A), false, (B), (short)0, (C), false, false)

// ---------------------------------------------------------------- routing ---

__global__ void moe_zero_counts(int* counts) {
    if (threadIdx.x < 16) counts[threadIdx.x] = 0;
}

__global__ void moe_router(const float* __restrict__ logits,
                           float* __restrict__ wk, int* __restrict__ tkidx,
                           int* __restrict__ counts) {
    int t = blockIdx.x * blockDim.x + threadIdx.x;
    if (t >= T_) return;
    float l[E_];
#pragma unroll
    for (int e = 0; e < E_; ++e) l[e] = logits[t * E_ + e];
    int i0 = 0; float v0 = l[0];
#pragma unroll
    for (int e = 1; e < E_; ++e) if (l[e] > v0) { v0 = l[e]; i0 = e; }
    int i1 = -1; float v1 = -3.4e38f;
#pragma unroll
    for (int e = 0; e < E_; ++e) if (e != i0 && l[e] > v1) { v1 = l[e]; i1 = e; }
    // softmax + top2 + renorm collapses to a 2-way logistic on the top-2 logits
    float w0 = __builtin_amdgcn_rcpf(1.0f + __expf(v1 - v0));
    wk[t * 2 + 0] = w0;
    wk[t * 2 + 1] = 1.0f - w0;
    tkidx[t * 2 + 0] = i0;
    tkidx[t * 2 + 1] = i1;
    atomicAdd(&counts[i0], 1);
    atomicAdd(&counts[i1], 1);
}

__global__ void moe_offsets(const int* __restrict__ counts, int* __restrict__ offs) {
    if (threadIdx.x == 0) {
        int acc = 0;
        offs[0] = 0;
        for (int e = 0; e < E_; ++e) {
            acc += ((counts[e] + TM_ - 1) / TM_) * TM_;  // tile-aligned
            offs[e + 1] = acc;
        }
    }
}

// Deterministic compaction: per-expert block scans tokens in order (LDS prefix sum).
__global__ void moe_fill(const int* __restrict__ tkidx, const int* __restrict__ offs,
                         int* __restrict__ pair_token, int* __restrict__ pair_dest) {
    const int e = blockIdx.x;
    const int tid = threadIdx.x;
    __shared__ int scan[256];
    int base = offs[e];
    for (int t0 = 0; t0 < T_; t0 += 256) {
        int t = t0 + tid;
        int k = -1;
        int e0 = tkidx[t * 2], e1 = tkidx[t * 2 + 1];
        if (e0 == e) k = 0; else if (e1 == e) k = 1;
        int flag = (k >= 0) ? 1 : 0;
        scan[tid] = flag;
        __syncthreads();
        for (int off = 1; off < 256; off <<= 1) {
            int v = (tid >= off) ? scan[tid - off] : 0;
            __syncthreads();
            scan[tid] += v;
            __syncthreads();
        }
        if (k >= 0) {
            int pos = base + scan[tid] - flag;
            pair_token[pos] = t;
            pair_dest[pos]  = t * 2 + k;
        }
        base += scan[255];
        __syncthreads();
    }
    // pad tail of this expert's tile-aligned region
    for (int p = base + tid; p < offs[e + 1]; p += 256) {
        pair_token[p] = 0;
        pair_dest[p]  = -1;
    }
}

// Map linear tile id -> (expert, local tile). Uniform across block.
__device__ __forceinline__ int map_tile(const int* counts, int tileId, int* localOut) {
    int e = -1, local = 0, acc = 0;
#pragma unroll
    for (int ee = 0; ee < E_; ++ee) {
        int te = (counts[ee] + TM_ - 1) / TM_;
        if (e < 0 && tileId < acc + te) { e = ee; local = tileId - acc; }
        acc += te;
    }
    *localOut = local;
    return e;
}

// -------------------------------------------------- GEMM1: silu(XW1^T)*XW3^T ---

__global__ void __launch_bounds__(256)
moe_gemm1(const float* __restrict__ X, const float* __restrict__ W1,
          const float* __restrict__ W3, const int* __restrict__ counts,
          const int* __restrict__ offs, const int* __restrict__ pair_token,
          __bf16* __restrict__ h) {
    const int tid = threadIdx.x;
    int local;
    const int e = map_tile(counts, blockIdx.x, &local);
    if (e < 0) return;
    const int rowBase = offs[e] + local * TM_;
    const int nBase = blockIdx.y * TN_;

    __shared__ __align__(16) __bf16 Xs[2][TM_][TK_];
    __shared__ __align__(16) __bf16 W1s[2][TN_][TK_];
    __shared__ __align__(16) __bf16 W3s[2][TN_][TK_];
    __shared__ int toks[TM_];
    if (tid < TM_) toks[tid] = pair_token[rowBase + tid];
    __syncthreads();

    // per-thread staging geometry
    const int xr = tid >> 3, xc = (tid & 7) * 4;    // X tile: 32x32, 4 elem/thread
    const int wr = tid >> 1, wc = (tid & 1) * 16;   // W tiles: 128x32, 16 elem/thread
    const size_t xbase = (size_t)toks[xr] * H_ + xc;
    const size_t wbase = ((size_t)e * I_ + nBase + wr) * H_ + wc;

    float4 xA, w1A[4], w3A[4];
    auto loadStage = [&](int kk) {
        xA = *(const float4*)(X + xbase + kk);
        const float* p1 = W1 + wbase + kk;
        const float* p3 = W3 + wbase + kk;
        __builtin_prefetch(p1 + TK_, 0, 3);  // near-cache prefetch of chunk after next
        __builtin_prefetch(p3 + TK_, 0, 3);
#pragma unroll
        for (int j = 0; j < 4; ++j) {
            w1A[j] = *(const float4*)(p1 + 4 * j);
            w3A[j] = *(const float4*)(p3 + 4 * j);
        }
    };
    auto storeStage = [&](int buf) {
        *(bf16x4*)&Xs[buf][xr][xc] = cvt4(xA);
#pragma unroll
        for (int j = 0; j < 4; ++j) {
            *(bf16x4*)&W1s[buf][wr][wc + 4 * j] = cvt4(w1A[j]);
            *(bf16x4*)&W3s[buf][wr][wc + 4 * j] = cvt4(w3A[j]);
        }
    };

    const int wave = tid >> 5, lane = tid & 31;
    const int mi = wave >> 2;            // 0..1 (M subtile)
    const int ni0 = (wave & 3) * 2;      // 0,2,4,6 (first of 2 N subtiles)
    const int lm = lane & 15;
    const int kb = (lane >> 4) * 8;

    const v8f vz = {0, 0, 0, 0, 0, 0, 0, 0};
    v8f acc1a = vz, acc1b = vz, acc3a = vz, acc3b = vz;

    loadStage(0);
    int cur = 0;
    for (int kk = 0; kk < H_; kk += TK_) {
        storeStage(cur);
        __syncthreads();                         // dscnt flush; loadcnt overlaps
        if (kk + TK_ < H_) loadStage(kk + TK_);  // prefetch next chunk into regs
        // issue ALL fragment loads first (distinct regs), then the 4 WMMAs,
        // so ds_load latency pipelines instead of 4x s_wait_dscnt 0 stalls
        v16bf af  = load_frag(&Xs[cur][0][0],  mi * 16 + lm, kb);
        v16bf b1a = load_frag(&W1s[cur][0][0], (ni0 + 0) * 16 + lm, kb);
        v16bf b3a = load_frag(&W3s[cur][0][0], (ni0 + 0) * 16 + lm, kb);
        v16bf b1b = load_frag(&W1s[cur][0][0], (ni0 + 1) * 16 + lm, kb);
        v16bf b3b = load_frag(&W3s[cur][0][0], (ni0 + 1) * 16 + lm, kb);
        acc1a = WMMA_BF16(af, b1a, acc1a);
        acc3a = WMMA_BF16(af, b3a, acc3a);
        acc1b = WMMA_BF16(af, b1b, acc1b);
        acc3b = WMMA_BF16(af, b3b, acc3b);
        cur ^= 1;
    }

    // fused SiLU-gate epilogue; D layout: VGPR i <-> M, lane <-> N
    const int mBase = (lane >> 4) * 8;
    const int pcolA = nBase + (ni0 + 0) * 16 + lm;
    const int pcolB = nBase + (ni0 + 1) * 16 + lm;
#pragma unroll
    for (int i = 0; i < 8; ++i) {
        const int prow = rowBase + mi * 16 + mBase + i;
        float ga = acc1a[i], ua = acc3a[i];
        float va = ga * __builtin_amdgcn_rcpf(1.0f + __expf(-ga)) * ua;
        h[(size_t)prow * I_ + pcolA] = (__bf16)va;
        float gb = acc1b[i], ub = acc3b[i];
        float vb = gb * __builtin_amdgcn_rcpf(1.0f + __expf(-gb)) * ub;
        h[(size_t)prow * I_ + pcolB] = (__bf16)vb;
    }
}

// ------------------------------------------------------ GEMM2: h @ W2^T ---

__global__ void __launch_bounds__(256)
moe_gemm2(const __bf16* __restrict__ h, const float* __restrict__ W2,
          const int* __restrict__ counts, const int* __restrict__ offs,
          const int* __restrict__ pair_dest, float* __restrict__ partial) {
    const int tid = threadIdx.x;
    int local;
    const int e = map_tile(counts, blockIdx.x, &local);
    if (e < 0) return;
    const int rowBase = offs[e] + local * TM_;
    const int nBase = blockIdx.y * TN_;   // H columns

    __shared__ __align__(16) __bf16 Hs[2][TM_][TK_];
    __shared__ __align__(16) __bf16 W2s[2][TN_][TK_];

    const int hr = tid >> 3, hc = (tid & 7) * 4;
    const int wr = tid >> 1, wc = (tid & 1) * 16;
    const size_t hbase  = (size_t)(rowBase + hr) * I_ + hc;
    const size_t w2base = ((size_t)e * H_ + nBase + wr) * I_ + wc;

    bf16x4 hA;
    float4 w2A[4];
    auto loadStage = [&](int kk) {
        hA = *(const bf16x4*)(h + hbase + kk);     // bf16 direct, 8B load
        const float* p2 = W2 + w2base + kk;
        __builtin_prefetch(p2 + TK_, 0, 3);
#pragma unroll
        for (int j = 0; j < 4; ++j) w2A[j] = *(const float4*)(p2 + 4 * j);
    };
    auto storeStage = [&](int buf) {
        *(bf16x4*)&Hs[buf][hr][hc] = hA;
#pragma unroll
        for (int j = 0; j < 4; ++j)
            *(bf16x4*)&W2s[buf][wr][wc + 4 * j] = cvt4(w2A[j]);
    };

    const int wave = tid >> 5, lane = tid & 31;
    const int mi = wave >> 2;
    const int ni0 = (wave & 3) * 2;
    const int lm = lane & 15;
    const int kb = (lane >> 4) * 8;

    const v8f vz = {0, 0, 0, 0, 0, 0, 0, 0};
    v8f accA = vz, accB = vz;

    loadStage(0);
    int cur = 0;
    for (int kk = 0; kk < I_; kk += TK_) {
        storeStage(cur);
        __syncthreads();
        if (kk + TK_ < I_) loadStage(kk + TK_);
        v16bf af  = load_frag(&Hs[cur][0][0],  mi * 16 + lm, kb);
        v16bf bfa = load_frag(&W2s[cur][0][0], (ni0 + 0) * 16 + lm, kb);
        v16bf bfb = load_frag(&W2s[cur][0][0], (ni0 + 1) * 16 + lm, kb);
        accA = WMMA_BF16(af, bfa, accA);
        accB = WMMA_BF16(af, bfb, accB);
        cur ^= 1;
    }

    // per-(token,k) partial slot -> deterministic, no float atomics
    const int mBase = (lane >> 4) * 8;
    const int pcolA = nBase + (ni0 + 0) * 16 + lm;
    const int pcolB = nBase + (ni0 + 1) * 16 + lm;
#pragma unroll
    for (int i = 0; i < 8; ++i) {
        const int prow = rowBase + mi * 16 + mBase + i;
        const int dest = pair_dest[prow];
        if (dest >= 0) {
            partial[(size_t)dest * H_ + pcolA] = accA[i];
            partial[(size_t)dest * H_ + pcolB] = accB[i];
        }
    }
}

// out[t] = w0 * partial[2t] + w1 * partial[2t+1]
__global__ void moe_combine(const float* __restrict__ partial,
                            const float* __restrict__ wk, float* __restrict__ out) {
    size_t idx = ((size_t)blockIdx.x * blockDim.x + threadIdx.x) * 4;
    int t = (int)(idx / H_);
    int c = (int)(idx % H_);
    float w0 = wk[t * 2], w1 = wk[t * 2 + 1];
    float4 p0 = *(const float4*)(partial + (size_t)(t * 2) * H_ + c);
    float4 p1 = *(const float4*)(partial + (size_t)(t * 2 + 1) * H_ + c);
    float4 o;
    o.x = w0 * p0.x + w1 * p1.x;
    o.y = w0 * p0.y + w1 * p1.y;
    o.z = w0 * p0.z + w1 * p1.z;
    o.w = w0 * p0.w + w1 * p1.w;
    *(float4*)(out + idx) = o;
}

// ----------------------------------------------------------------- launch ---

extern "C" void kernel_launch(void* const* d_in, const int* in_sizes, int n_in,
                              void* d_out, int out_size, void* d_ws, size_t ws_size,
                              hipStream_t stream) {
    (void)in_sizes; (void)n_in; (void)out_size; (void)ws_size;
    const float* X  = (const float*)d_in[0];   // [T,H]
    const float* lg = (const float*)d_in[1];   // [T,E]
    const float* W1 = (const float*)d_in[2];   // [E,I,H]
    const float* W3 = (const float*)d_in[3];   // [E,I,H]
    const float* W2 = (const float*)d_in[4];   // [E,H,I]
    float* out = (float*)d_out;                // [T,H]
    char* ws = (char*)d_ws;

    // workspace layout (~161 MB total)
    size_t o = 0;
    int*    counts = (int*)(ws + o);    o += 256;
    int*    offs   = (int*)(ws + o);    o += 256;
    float*  wk     = (float*)(ws + o);  o += (size_t)T_ * 2 * sizeof(float);
    int*    tkidx  = (int*)(ws + o);    o += (size_t)T_ * 2 * sizeof(int);
    int*    ptok   = (int*)(ws + o);    o += (((size_t)CAP_ * 4) + 255) / 256 * 256;
    int*    pdest  = (int*)(ws + o);    o += (((size_t)CAP_ * 4) + 255) / 256 * 256;
    __bf16* hbuf   = (__bf16*)(ws + o); o += (((size_t)CAP_ * I_ * 2) + 255) / 256 * 256;
    float*  part   = (float*)(ws + o);  o += (size_t)NPAIR_ * H_ * sizeof(float);

    moe_zero_counts<<<1, 32, 0, stream>>>(counts);
    moe_router<<<T_ / 256, 256, 0, stream>>>(lg, wk, tkidx, counts);
    moe_offsets<<<1, 32, 0, stream>>>(counts, offs);
    moe_fill<<<E_, 256, 0, stream>>>(tkidx, offs, ptok, pdest);
    moe_gemm1<<<dim3(MAX_TILES_, I_ / TN_), 256, 0, stream>>>(X, W1, W3, counts, offs, ptok, hbuf);
    moe_gemm2<<<dim3(MAX_TILES_, H_ / TN_), 256, 0, stream>>>(hbuf, W2, counts, offs, pdest, part);
    moe_combine<<<(T_ * H_ / 4) / 256, 256, 0, stream>>>(part, wk, out);
}
